// AttentiveFPPredictor_14044543058378
// MI455X (gfx1250) — compile-verified, matrix operand order checked by reading, not run on hardware
//
#include <hip/hip_runtime.h>
#include <hip/hip_bf16.h>
#include <cstddef>

// Problem sizes (fixed by the reference).
#define V_NODES  100000
#define V_PAD    100032    // round up to 64 rows: GEMM epilogue stores unconditionally
#define E_EDGES  200000    // already a multiple of 64
#define G_GRAPHS 4096      // already a multiple of 64
#define F_DIM    256
#define NODE_IN  64
#define EDGE_IN  16
#define CE_PAD   96        // concat(node 64, edge 16) padded 80 -> 96 (K % 32 == 0)

// ---------------------------------------------------------------------------
// WMMA types (CDNA5 / gfx1250, wave32)
// ---------------------------------------------------------------------------
typedef __bf16 bf16_t;
typedef bf16_t v16bf __attribute__((ext_vector_type(16)));
typedef float  v8f   __attribute__((ext_vector_type(8)));

union FragBF16 {
  v16bf        v;
  unsigned int u[8];
};

__device__ inline unsigned short f32_to_bf16_rne(float f) {
  unsigned int u = __float_as_uint(f);
  unsigned int r = u + 0x7FFFu + ((u >> 16) & 1u);   // round-to-nearest-even
  return (unsigned short)(r >> 16);
}

__device__ inline float lrelu01(float x) { return x > 0.f ? x : 0.01f * x; }
__device__ inline float sigmf(float x)   { return 1.f / (1.f + expf(-x)); }
__device__ inline float eluf(float x)    { return x > 0.f ? x : (expf(x) - 1.f); }

// monotone float <-> uint encoding for atomicMax-based segment max
__device__ inline unsigned int f2ord(float f) {
  unsigned int u = __float_as_uint(f);
  return (u & 0x80000000u) ? ~u : (u | 0x80000000u);
}
__device__ inline float ord2f(unsigned int u) {
  unsigned int v = (u & 0x80000000u) ? (u & 0x7FFFFFFFu) : ~u;
  return __uint_as_float(v);
}

// ---------------------------------------------------------------------------
// Generic GEMM:  Y[Mpad,N] = act(X[M,K] @ W[N,K]^T + bias[N])
// f32 global data, bf16 WMMA compute (v_wmma_f32_16x16x32_bf16), f32 accum.
// Block = 128 threads (4 waves), tile 64x64, K step 32, LDS staged.
// REQUIRES: K % 32 == 0, N % 64 == 0, Y has ceil(M/64)*64 rows allocated.
// A-rows are clamped for loads (branchless); stores are unconditional into
// the padded output (padded rows hold garbage and are never read).
// ---------------------------------------------------------------------------
#define BM 64
#define BN 64
#define BK 32
#define LDS_STRIDE (BK + 2)   // 34 ushorts = 68 B = 17 banks -> conflict free

__global__ __launch_bounds__(128)
void gemm_bias_act_kernel(const float* __restrict__ X, const float* __restrict__ W,
                          const float* __restrict__ bias, float* __restrict__ Y,
                          int M, int N, int K, int act)
{
  __shared__ __align__(16) unsigned short As[BM][LDS_STRIDE];
  __shared__ __align__(16) unsigned short Bs[BN][LDS_STRIDE];

  const int tid  = threadIdx.x;
  const int lane = tid & 31;
  const int wv   = tid >> 5;              // wave 0..3
  const int m0g  = blockIdx.y * BM;
  const int n0g  = blockIdx.x * BN;
  const int m0w  = (wv >> 1) * 32;        // wave's tile origin in block tile
  const int n0w  = (wv & 1) * 32;

  v8f acc[2][2];
  for (int i = 0; i < 2; ++i)
    for (int j = 0; j < 2; ++j)
      for (int r = 0; r < 8; ++r) acc[i][j][r] = 0.f;

  const int r0 = tid >> 3;                // 0..15
  const int c0 = (tid & 7) * 4;           // 0,4,...,28
  const int nK = K / BK;

  for (int kt = 0; kt < nK; ++kt) {
    const int gk = kt * BK + c0;
    // -------- stage X tile (BM x BK) and W tile (BN x BK) into LDS as bf16
    // branchless: X rows clamped to M-1 (padded rows never read downstream)
    for (int p = 0; p < 4; ++p) {
      const int r = r0 + p * 16;
      int grow = m0g + r;
      grow = grow < M ? grow : (M - 1);
      const float4 xv = *(const float4*)(X + (size_t)grow * K + gk);
      As[r][c0 + 0] = f32_to_bf16_rne(xv.x);
      As[r][c0 + 1] = f32_to_bf16_rne(xv.y);
      As[r][c0 + 2] = f32_to_bf16_rne(xv.z);
      As[r][c0 + 3] = f32_to_bf16_rne(xv.w);
      const float4 wvv = *(const float4*)(W + (size_t)(n0g + r) * K + gk);
      Bs[r][c0 + 0] = f32_to_bf16_rne(wvv.x);
      Bs[r][c0 + 1] = f32_to_bf16_rne(wvv.y);
      Bs[r][c0 + 2] = f32_to_bf16_rne(wvv.z);
      Bs[r][c0 + 3] = f32_to_bf16_rne(wvv.w);
    }
    __syncthreads();

    // -------- build fragments per ISA 16-bit A/B layout and issue WMMAs
    const int rsel = lane & 15;
    const int koff = (lane >> 4) * 8;
    FragBF16 afrag[2], bfrag[2];
    for (int t = 0; t < 2; ++t) {
      const int ar = m0w + t * 16 + rsel;
      const int br = n0w + t * 16 + rsel;
      for (int i = 0; i < 4; ++i) {
        afrag[t].u[i]     = *(const unsigned int*)&As[ar][koff + 2 * i];
        afrag[t].u[4 + i] = *(const unsigned int*)&As[ar][16 + koff + 2 * i];
        bfrag[t].u[i]     = *(const unsigned int*)&Bs[br][koff + 2 * i];
        bfrag[t].u[4 + i] = *(const unsigned int*)&Bs[br][16 + koff + 2 * i];
      }
    }
    for (int tm = 0; tm < 2; ++tm)
      for (int tn = 0; tn < 2; ++tn)
        acc[tm][tn] = __builtin_amdgcn_wmma_f32_16x16x32_bf16(
            false, afrag[tm].v, false, bfrag[tn].v,
            (short)0, acc[tm][tn], false, false);
    __syncthreads();
  }

  // -------- epilogue: bias + activation, unconditional (padded) stores
  const int rbase = (lane >> 4) * 8;
  for (int tm = 0; tm < 2; ++tm) {
    for (int tn = 0; tn < 2; ++tn) {
      const int ncol = n0g + n0w + tn * 16 + (lane & 15);
      const float bv = bias[ncol];
      for (int r = 0; r < 8; ++r) {
        const int mrow = m0g + m0w + tm * 16 + rbase + r;
        float v = acc[tm][tn][r] + bv;
        if (act == 1) v = lrelu01(v);
        Y[(size_t)mrow * N + ncol] = v;
      }
    }
  }
}

// ---------------------------------------------------------------------------
// Elementwise / attention / aggregation kernels
// ---------------------------------------------------------------------------
// ce[e, 0:64] = node_feats[src[e]], ce[e, 64:80] = edge_feats[e], ce[e, 80:96] = 0
__global__ void build_edge_concat_kernel(const float* __restrict__ nf,
                                         const float* __restrict__ ef,
                                         const int* __restrict__ src,
                                         float* __restrict__ out)
{
  const int e = blockIdx.x;
  const int t = threadIdx.x;
  if (t < CE_PAD) {
    float v = 0.f;
    if (t < NODE_IN)                v = nf[(size_t)src[e] * NODE_IN + t];
    else if (t < NODE_IN + EDGE_IN) v = ef[(size_t)e * EDGE_IN + (t - NODE_IN)];
    out[(size_t)e * CE_PAD + t] = v;
  }
}

// zero-padded copy of proj_edge1 weight: [256,80] -> [256,96]
__global__ void pad_weight_kernel(const float* __restrict__ w, float* __restrict__ out)
{
  const int i = blockIdx.x * blockDim.x + threadIdx.x;   // over 256*96
  if (i < F_DIM * CE_PAD) {
    const int n = i / CE_PAD, k = i % CE_PAD;
    out[i] = (k < NODE_IN + EDGE_IN) ? w[(size_t)n * (NODE_IN + EDGE_IN) + k] : 0.f;
  }
}

// logit[i] = lrelu( w[0:256].L[rowL] + w[256:512].R[rowR] + b ), one wave/row
__global__ __launch_bounds__(256)
void pair_logits_kernel(const float* __restrict__ w, const float* __restrict__ bias,
                        const float* __restrict__ L, const int* __restrict__ idxL,
                        const float* __restrict__ R, const int* __restrict__ idxR,
                        int reluL, float* __restrict__ out, int n)
{
  const int e = blockIdx.x * 8 + (threadIdx.x >> 5);
  if (e >= n) return;
  const int lane = threadIdx.x & 31;
  const int rl = idxL ? idxL[e] : e;
  const int rr = idxR ? idxR[e] : e;
  const float* pl = L + (size_t)rl * F_DIM;
  const float* pr = R + (size_t)rr * F_DIM;
  float s = 0.f;
  for (int i = 0; i < 8; ++i) {
    const int f = i * 32 + lane;
    float lv = pl[f];
    if (reluL) lv = fmaxf(lv, 0.f);
    s += w[f] * lv + w[F_DIM + f] * pr[f];
  }
  for (int off = 16; off > 0; off >>= 1) s += __shfl_down(s, off, 32);
  if (lane == 0) out[e] = lrelu01(s + bias[0]);
}

__global__ void fill_u32_kernel(unsigned int* p, unsigned int v, int n) {
  int i = blockIdx.x * blockDim.x + threadIdx.x;
  if (i < n) p[i] = v;
}
__global__ void fill_f32_kernel(float* p, float v, size_t n) {
  size_t i = (size_t)blockIdx.x * blockDim.x + threadIdx.x;
  if (i < n) p[i] = v;
}
// vectorized variants: n4 = count of float4 elements
__global__ void fill4_kernel(float4* p, size_t n4) {
  size_t i = (size_t)blockIdx.x * blockDim.x + threadIdx.x;
  if (i < n4) p[i] = make_float4(0.f, 0.f, 0.f, 0.f);
}
__global__ void copy4_kernel(const float4* __restrict__ in, float4* __restrict__ out, size_t n4) {
  size_t i = (size_t)blockIdx.x * blockDim.x + threadIdx.x;
  if (i < n4) out[i] = in[i];
}
__global__ void elu4_kernel(const float4* __restrict__ in, float4* __restrict__ out, size_t n4) {
  size_t i = (size_t)blockIdx.x * blockDim.x + threadIdx.x;
  if (i < n4) {
    float4 v = in[i];
    v.x = eluf(v.x); v.y = eluf(v.y); v.z = eluf(v.z); v.w = eluf(v.w);
    out[i] = v;
  }
}

__global__ void seg_max_kernel(const float* __restrict__ lg, const int* __restrict__ seg,
                               unsigned int* __restrict__ mx, int n) {
  int i = blockIdx.x * blockDim.x + threadIdx.x;
  if (i < n) atomicMax(&mx[seg[i]], f2ord(lg[i]));
}
__global__ void seg_expsum_kernel(const float* __restrict__ lg, const int* __restrict__ seg,
                                  const unsigned int* __restrict__ mx,
                                  float* __restrict__ sm, int n) {
  int i = blockIdx.x * blockDim.x + threadIdx.x;
  if (i < n) atomicAdd(&sm[seg[i]], expf(lg[i] - ord2f(mx[seg[i]])));
}
__global__ void seg_norm_kernel(const float* __restrict__ lg, const int* __restrict__ seg,
                                const unsigned int* __restrict__ mx,
                                const float* __restrict__ sm, float* __restrict__ a, int n) {
  int i = blockIdx.x * blockDim.x + threadIdx.x;
  if (i < n) a[i] = expf(lg[i] - ord2f(mx[seg[i]])) / sm[seg[i]];
}

// out[dst[e]] += scale * vals[(gather? gather[e] : e)]   (256 threads = F_DIM)
__global__ __launch_bounds__(256)
void scatter_scale_add_kernel(const float* __restrict__ vals, const int* __restrict__ gather,
                              const float* __restrict__ a, const int* __restrict__ dst,
                              float* __restrict__ out)
{
  const int e = blockIdx.x;
  const int f = threadIdx.x;
  const int r = gather ? gather[e] : e;
  const float s = a ? a[e] : 1.f;
  atomicAdd(&out[(size_t)dst[e] * F_DIM + f], s * vals[(size_t)r * F_DIM + f]);
}

// torch GRUCell combine + ReLU (float4): gi/gh [rows,3F] (r,z,n), h/out [rows,F]
__global__ __launch_bounds__(256)
void gru_combine_kernel(const float* __restrict__ gi, const float* __restrict__ gh,
                        const float* __restrict__ h, float* __restrict__ out, int rows)
{
  const int Q = F_DIM / 4;   // 64 float4 per row
  size_t idx = (size_t)blockIdx.x * blockDim.x + threadIdx.x;
  if (idx >= (size_t)rows * Q) return;
  const int row = (int)(idx / Q);
  const int c   = (int)(idx % Q);
  const float4* pi = (const float4*)(gi + (size_t)row * 3 * F_DIM);
  const float4* ph = (const float4*)(gh + (size_t)row * 3 * F_DIM);
  const float4 ir = pi[c],         hr = ph[c];
  const float4 iz = pi[Q + c],     hz = ph[Q + c];
  const float4 in_ = pi[2 * Q + c], hn = ph[2 * Q + c];
  const float4 hv = ((const float4*)(h + (size_t)row * F_DIM))[c];
  float4 o;
  {
    const float r0 = sigmf(ir.x + hr.x), z0 = sigmf(iz.x + hz.x);
    o.x = fmaxf((1.f - z0) * tanhf(in_.x + r0 * hn.x) + z0 * hv.x, 0.f);
    const float r1 = sigmf(ir.y + hr.y), z1 = sigmf(iz.y + hz.y);
    o.y = fmaxf((1.f - z1) * tanhf(in_.y + r1 * hn.y) + z1 * hv.y, 0.f);
    const float r2 = sigmf(ir.z + hr.z), z2 = sigmf(iz.z + hz.z);
    o.z = fmaxf((1.f - z2) * tanhf(in_.z + r2 * hn.z) + z2 * hv.z, 0.f);
    const float r3 = sigmf(ir.w + hr.w), z3 = sigmf(iz.w + hz.w);
    o.w = fmaxf((1.f - z3) * tanhf(in_.w + r3 * hn.w) + z3 * hv.w, 0.f);
  }
  ((float4*)(out + (size_t)row * F_DIM))[c] = o;
}

// ---------------------------------------------------------------------------
// Host orchestration
// ---------------------------------------------------------------------------
extern "C" void kernel_launch(void* const* d_in, const int* in_sizes, int n_in,
                              void* d_out, int out_size, void* d_ws, size_t ws_size,
                              hipStream_t stream)
{
  (void)in_sizes; (void)n_in; (void)out_size; (void)ws_size;

  // Pytree flattening assumed to follow setup_inputs() insertion order,
  // recursing through dicts (insertion order) and lists (in order).
  const float* node_feats = (const float*)d_in[0];
  const float* edge_feats = (const float*)d_in[1];
  #define PF(i) ((const float*)d_in[(i)])
  // params:
  // 2,3   proj_node.W/b     4,5  proj_edge1.W/b   6,7  proj_edge2.W/b
  // 8,9   edge_transform    10..13 gru0 Wih/Whh/bih/bhh
  // layers l: base 14+8l : proj_edge.W/b, proj_node.W/b, gru Wih/Whh/bih/bhh
  // readout t: base 30+8t : compute_logits.W/b, project_nodes.W/b, gru ...
  const int* src = (const int*)d_in[46];
  const int* dst = (const int*)d_in[47];
  const int* gid = (const int*)d_in[48];

  // workspace layout (floats); V-row GEMM outputs padded to V_PAD rows
  float* ws = (float*)d_ws;
  size_t o = 0;
  float* hv   = ws + o; o += (size_t)V_PAD * F_DIM;            // GetContext node proj
  float* he1  = ws + o; o += (size_t)E_EDGES * F_DIM;          // edge hidden
  float* ht   = ws + o; o += (size_t)E_EDGES * F_DIM;          // transformed edges / node proj
  float* cbuf = ws + o; o += (size_t)V_PAD * F_DIM;            // context accum (also g_repr)
  float* xe   = ws + o; o += (size_t)V_PAD * F_DIM;            // elu(c)
  float* gi   = ws + o; o += (size_t)V_PAD * 3 * F_DIM;        // GRU input gates
  float* gh   = ws + o; o += (size_t)V_PAD * 3 * F_DIM;        // GRU hidden gates
  float* node = ws + o; o += (size_t)V_PAD * F_DIM;            // node state
  float* ce   = ws + o; o += (size_t)E_EDGES * CE_PAD;         // padded edge concat
  float* wpad = ws + o; o += (size_t)F_DIM * CE_PAD;           // padded proj_edge1.W
  float* lg   = ws + o; o += E_EDGES;                          // logits
  float* av   = ws + o; o += E_EDGES;                          // attention weights
  unsigned int* mx = (unsigned int*)(ws + o); o += V_NODES;    // segment max (encoded)
  float* sm   = ws + o; o += V_NODES;                          // segment expsum
  float* gf   = ws + o; o += (size_t)G_GRAPHS * F_DIM;         // graph feats

  auto gemm = [&](const float* X, const float* Wp, const float* bp, float* Y,
                  int M, int N, int K, int act) {
    dim3 grid(N / BN, (M + BM - 1) / BM);
    gemm_bias_act_kernel<<<grid, 128, 0, stream>>>(X, Wp, bp, Y, M, N, K, act);
  };
  auto zero = [&](float* p, size_t n) {   // n % 4 == 0
    fill4_kernel<<<(unsigned)((n / 4 + 255) / 256), 256, 0, stream>>>((float4*)p, n / 4);
  };
  auto elu = [&](const float* in, float* out2, size_t n) {  // n % 4 == 0
    elu4_kernel<<<(unsigned)((n / 4 + 255) / 256), 256, 0, stream>>>((const float4*)in, (float4*)out2, n / 4);
  };
  auto seg_softmax = [&](const float* lgp, const int* seg, int n, int nseg, float* ap) {
    fill_u32_kernel<<<(nseg + 255) / 256, 256, 0, stream>>>(mx, 0u, nseg);
    fill_f32_kernel<<<(nseg + 255) / 256, 256, 0, stream>>>(sm, 0.f, (size_t)nseg);
    seg_max_kernel<<<(n + 255) / 256, 256, 0, stream>>>(lgp, seg, mx, n);
    seg_expsum_kernel<<<(n + 255) / 256, 256, 0, stream>>>(lgp, seg, mx, sm, n);
    seg_norm_kernel<<<(n + 255) / 256, 256, 0, stream>>>(lgp, seg, mx, sm, ap, n);
  };

  const size_t VF = (size_t)V_NODES * F_DIM;
  const size_t GF = (size_t)G_GRAPHS * F_DIM;

  // ================= GetContext =================
  gemm(node_feats, PF(2), PF(3), hv, V_NODES, F_DIM, NODE_IN, 1);          // hv_new = lrelu(proj_node)
  build_edge_concat_kernel<<<E_EDGES, 128, 0, stream>>>(node_feats, edge_feats, src, ce);
  pad_weight_kernel<<<(F_DIM * CE_PAD + 255) / 256, 256, 0, stream>>>(PF(4), wpad);
  gemm(ce, wpad, PF(5), he1, E_EDGES, F_DIM, CE_PAD, 1);                   // he1 = lrelu(proj_edge1)
  pair_logits_kernel<<<(E_EDGES + 7) / 8, 256, 0, stream>>>(PF(6), PF(7), hv, dst, he1, nullptr, 0, lg, E_EDGES);
  seg_softmax(lg, dst, E_EDGES, V_NODES, av);
  gemm(he1, PF(8), PF(9), ht, E_EDGES, F_DIM, F_DIM, 0);                   // edge_transform(he1)
  zero(cbuf, VF);
  scatter_scale_add_kernel<<<E_EDGES, 256, 0, stream>>>(ht, nullptr, av, dst, cbuf);
  elu(cbuf, xe, VF);
  gemm(xe, PF(10), PF(12), gi, V_NODES, 3 * F_DIM, F_DIM, 0);              // elu(c) @ Wih^T + bih
  gemm(hv, PF(11), PF(13), gh, V_NODES, 3 * F_DIM, F_DIM, 0);              // hv @ Whh^T + bhh
  gru_combine_kernel<<<(unsigned)((VF / 4 + 255) / 256), 256, 0, stream>>>(gi, gh, hv, node, V_NODES);

  // ================= GNN layers =================
  for (int l = 0; l < 2; ++l) {
    const int pb = 14 + 8 * l;
    pair_logits_kernel<<<(E_EDGES + 7) / 8, 256, 0, stream>>>(PF(pb + 0), PF(pb + 1), node, dst, node, src, 0, lg, E_EDGES);
    seg_softmax(lg, dst, E_EDGES, V_NODES, av);
    gemm(node, PF(pb + 2), PF(pb + 3), ht, V_NODES, F_DIM, F_DIM, 0);      // hp = proj_node(node)
    zero(cbuf, VF);
    scatter_scale_add_kernel<<<E_EDGES, 256, 0, stream>>>(ht, src, av, dst, cbuf);  // c[dst] += a*hp[src]
    elu(cbuf, xe, VF);
    gemm(xe, PF(pb + 4), PF(pb + 6), gi, V_NODES, 3 * F_DIM, F_DIM, 0);
    gemm(node, PF(pb + 5), PF(pb + 7), gh, V_NODES, 3 * F_DIM, F_DIM, 0);
    gru_combine_kernel<<<(unsigned)((VF / 4 + 255) / 256), 256, 0, stream>>>(gi, gh, node, node, V_NODES);
  }

  // ================= Readout =================
  zero(gf, GF);
  scatter_scale_add_kernel<<<V_NODES, 256, 0, stream>>>(node, nullptr, nullptr, gid, gf);  // segment_sum
  for (int t = 0; t < 2; ++t) {
    const int pb = 30 + 8 * t;
    pair_logits_kernel<<<(V_NODES + 7) / 8, 256, 0, stream>>>(PF(pb + 0), PF(pb + 1), gf, gid, node, nullptr, 1, lg, V_NODES);
    seg_softmax(lg, gid, V_NODES, G_GRAPHS, av);
    gemm(node, PF(pb + 2), PF(pb + 3), ht, V_NODES, F_DIM, F_DIM, 0);      // hv = project_nodes(node)
    zero(cbuf, GF);
    scatter_scale_add_kernel<<<V_NODES, 256, 0, stream>>>(ht, nullptr, av, gid, cbuf);     // g_repr
    elu(cbuf, xe, GF);
    gemm(xe, PF(pb + 4), PF(pb + 6), gi, G_GRAPHS, 3 * F_DIM, F_DIM, 0);
    gemm(gf, PF(pb + 5), PF(pb + 7), gh, G_GRAPHS, 3 * F_DIM, F_DIM, 0);
    gru_combine_kernel<<<(unsigned)((GF / 4 + 255) / 256), 256, 0, stream>>>(gi, gh, gf, gf, G_GRAPHS);
  }

  copy4_kernel<<<(unsigned)((GF / 4 + 255) / 256), 256, 0, stream>>>((const float4*)gf, (float4*)d_out, GF / 4);
  #undef PF
}
